// HolographicMemory_52080773431950
// MI455X (gfx1250) — matrix operand order repaired and checked by reading, not compiled, not run on hardware
//
#include <hip/hip_runtime.h>
#include <hip/hip_bf16.h>

#define BB 64
#define SS 2048
#define DD 1024
#define HH 16
#define HD 64
#define CHUNK 64
#define NCHUNK (SS / CHUNK) /* 32 */

typedef __bf16 v16bf __attribute__((ext_vector_type(16)));
typedef __bf16 v8bf  __attribute__((ext_vector_type(8)));
typedef __bf16 v4bf  __attribute__((ext_vector_type(4)));
typedef float  v8f   __attribute__((ext_vector_type(8)));

static __device__ __forceinline__ v8f wmma_bf16(v16bf a, v16bf b, v8f c) {
  // D = A(16x32 bf16) x B(32x16 bf16) + C(16x16 f32)
  return __builtin_amdgcn_wmma_f32_16x16x32_bf16(false, a, false, b, (short)0, c,
                                                 false, false);
}

// Build 16-bit A fragment from an LDS row: contiguous run [kb..kb+16)
static __device__ __forceinline__ v16bf load_a_frag_lds(const __bf16* p) {
  v8bf lo = *(const v8bf*)p;
  v8bf hi = *(const v8bf*)(p + 8);
  v16bf a;
#pragma unroll
  for (int i = 0; i < 8; ++i) { a[i] = lo[i]; a[i + 8] = hi[i]; }
  return a;
}

// ---------------------------------------------------------------------------
// Kernel 0: convert Wk/Wv (f32 row-major DxD) into WMMA-B-fragment-ordered bf16
// fragment id f = ((h*32 + ktile)*4 + ntile); per lane 16 contiguous bf16:
//   col n = ntile*16 + (lane&15) (+ h*64),  K = ktile*32 + ((lane<16)?0:16) + 0..15
// ---------------------------------------------------------------------------
__global__ void prep_wb(const float* __restrict__ Wk, const float* __restrict__ Wv,
                        __bf16* __restrict__ wkf, __bf16* __restrict__ wvf) {
  int t    = blockIdx.x * blockDim.x + threadIdx.x; // 0..65535
  int lane = t & 31;
  int frag = t >> 5;                 // 0..2047
  int nt    = frag & 3;
  int ktile = (frag >> 2) & 31;
  int h     = frag >> 7;
  int nrow  = h * HD + nt * 16 + (lane & 15);
  int kb    = ktile * 32 + ((lane < 16) ? 0 : 16);
  const float* sk = Wk + (size_t)nrow * DD + kb;
  const float* sv = Wv + (size_t)nrow * DD + kb;
  __bf16* dk = wkf + (size_t)frag * 512 + lane * 16;
  __bf16* dv = wvf + (size_t)frag * 512 + lane * 16;
#pragma unroll
  for (int j = 0; j < 16; ++j) { dk[j] = (__bf16)sk[j]; dv[j] = (__bf16)sv[j]; }
}

// ---------------------------------------------------------------------------
// Generic small GEMM: C(64 x N) = [A0 | A1](64 x K) @ W^T(K x N) + bias
// W is row-major (N x K). bf16 WMMA, f32 accumulate.
// grid: (N/128, 4); block 256 (8 waves, one 16x16 C tile each).
// ---------------------------------------------------------------------------
__global__ void gemm_bias(const float* __restrict__ A0, const float* __restrict__ A1,
                          int K0, int K, const float* __restrict__ W,
                          const float* __restrict__ bias, float* __restrict__ C, int N) {
  int wave = threadIdx.x >> 5, lane = threadIdx.x & 31;
  int mtile = blockIdx.y;
  int ntile = blockIdx.x * 8 + wave;
  int arow  = mtile * 16 + (lane & 15);
  int nrow  = ntile * 16 + (lane & 15);
  int asel  = (lane < 16) ? 0 : 16;  // A-fragment K run
  int bsel  = (lane < 16) ? 0 : 16;  // B-fragment K run
  v8f c = {};
  for (int kt = 0; kt < K; kt += 32) {
    const float* ap; int koff;
    if (kt < K0) { ap = A0 + (size_t)arow * K0;        koff = kt; }
    else         { ap = A1 + (size_t)arow * (K - K0);  koff = kt - K0; }
    int kb = koff + asel;
    v16bf a, w;
#pragma unroll
    for (int j = 0; j < 8; ++j) {
      a[j]     = (__bf16)ap[kb + j];
      a[j + 8] = (__bf16)ap[kb + 8 + j];
    }
    const float* wp = W + (size_t)nrow * K + kt + bsel;
#pragma unroll
    for (int j = 0; j < 16; ++j) w[j] = (__bf16)wp[j];
    c = wmma_bf16(a, w, c);
  }
  int mbase = mtile * 16 + ((lane < 16) ? 0 : 8);
  int col   = ntile * 16 + (lane & 15);
  float bbv = bias[col];
#pragma unroll
  for (int r = 0; r < 8; ++r) C[(size_t)(mbase + r) * N + col] = c[r] + bbv;
}

// ---------------------------------------------------------------------------
// Fused K/V projection + scores + partial softmax + P*V for one (batch, chunk).
// memory chunk (64x1024) lives in LDS as bf16; K/V computed with WMMA.
// Wave mapping: wave = (matsel<<2) | ntile. Each wave owns one 16-col output
// tile of K (matsel=0) or V (matsel=1) across all four 16-row M tiles, so each
// Wk/Wv fragment is loaded exactly once per block per k-step.
// ---------------------------------------------------------------------------
__global__ void attn_chunk(const float* __restrict__ mem, const float* __restrict__ Qws,
                           const __bf16* __restrict__ wkf, const __bf16* __restrict__ wvf,
                           const float* __restrict__ bk, const float* __restrict__ bv,
                           const int* __restrict__ mcount,
                           float* __restrict__ part_m, float* __restrict__ part_l,
                           float* __restrict__ part_acc) {
  extern __shared__ char smem_raw[];
  __bf16* memb = (__bf16*)smem_raw;                 // [64][1024] bf16 : 128 KB
  __bf16* Kl   = memb + CHUNK * DD;                 // [64][64]
  __bf16* Vl   = Kl + CHUNK * HD;                   // [64][64]
  float*  qsh  = (float*)(Vl + CHUNK * HD);         // [64]
  float*  bksh = qsh + 64;                          // [64]
  float*  bvsh = bksh + 64;                         // [64]
  float*  psh  = bvsh + 64;                         // [64]
  float*  red  = psh + 64;                          // [8]

  int c = blockIdx.x, b = blockIdx.y;
  int tid = threadIdx.x;
  int mc = mcount[0];
  int nvalid = mc < SS ? mc : SS; if (nvalid < 0) nvalid = 0;
  int s0 = c * CHUNK;
  int rows = nvalid - s0;
  rows = rows < 0 ? 0 : (rows > CHUNK ? CHUNK : rows);

  if (rows == 0) { // empty chunk: neutral partials
    for (int h = 0; h < HH; ++h) {
      int idx = (b * HH + h) * NCHUNK + c;
      if (tid == 0) { part_m[idx] = -3e38f; part_l[idx] = 0.f; }
      if (tid < HD) part_acc[(size_t)idx * HD + tid] = 0.f;
    }
    return;
  }

  // Phase 0: stage memory chunk into LDS as bf16, zero-fill invalid rows.
  const float* msrc = mem + ((size_t)b * SS + s0) * DD;
  for (int i = tid * 4; i < CHUNK * DD; i += blockDim.x * 4) {
    int r = i >> 10;
    float4 v;
    if (r < rows) v = *(const float4*)(msrc + i);
    else          v = make_float4(0.f, 0.f, 0.f, 0.f);
    v4bf o; o[0] = (__bf16)v.x; o[1] = (__bf16)v.y; o[2] = (__bf16)v.z; o[3] = (__bf16)v.w;
    *(v4bf*)(memb + i) = o;
  }
  __syncthreads();

  int wave   = tid >> 5, lane = tid & 31;
  int ntile  = wave & 3;   // output 16-col tile
  int matsel = wave >> 2;  // 0 -> K, 1 -> V
  const __bf16* wf = matsel ? wvf : wkf;
  int asel  = (lane < 16) ? 0 : 16;
  int lrow  = lane & 15;
  int rbase = (lane < 16) ? 0 : 8;
  int col   = ntile * 16 + lrow;

  for (int h = 0; h < HH; ++h) {
    if (tid < HD) {
      qsh[tid]  = Qws[(size_t)b * DD + h * HD + tid];
      bksh[tid] = bk[h * HD + tid];
      bvsh[tid] = bv[h * HD + tid];
    }
    __syncthreads();

    v8f acc0 = {}, acc1 = {}, acc2 = {}, acc3 = {};
    for (int ktile = 0; ktile < 32; ++ktile) {
      size_t f = ((size_t)(h * 32 + ktile) * 4 + ntile) * 512 + lane * 16;
      v16bf bfrag = *(const v16bf*)(wf + f);
      __builtin_prefetch(wf + f + 4 * 512, 0, 0); // next k-step's fragment
      const __bf16* abase = memb + ktile * 32 + asel;
      v16bf a0 = load_a_frag_lds(abase + (size_t)(lrow)      * DD);
      acc0 = wmma_bf16(a0, bfrag, acc0);
      v16bf a1 = load_a_frag_lds(abase + (size_t)(16 + lrow) * DD);
      acc1 = wmma_bf16(a1, bfrag, acc1);
      v16bf a2 = load_a_frag_lds(abase + (size_t)(32 + lrow) * DD);
      acc2 = wmma_bf16(a2, bfrag, acc2);
      v16bf a3 = load_a_frag_lds(abase + (size_t)(48 + lrow) * DD);
      acc3 = wmma_bf16(a3, bfrag, acc3);
    }

    // Spill K or V column tile (+bias) to LDS for the softmax/PV stage.
    {
      __bf16* dst = matsel ? Vl : Kl;
      const float* bsh = matsel ? bvsh : bksh;
      float bc = bsh[col];
#pragma unroll
      for (int r = 0; r < 8; ++r) dst[(0 * 16 + rbase + r) * HD + col] = (__bf16)(acc0[r] + bc);
#pragma unroll
      for (int r = 0; r < 8; ++r) dst[(1 * 16 + rbase + r) * HD + col] = (__bf16)(acc1[r] + bc);
#pragma unroll
      for (int r = 0; r < 8; ++r) dst[(2 * 16 + rbase + r) * HD + col] = (__bf16)(acc2[r] + bc);
#pragma unroll
      for (int r = 0; r < 8; ++r) dst[(3 * 16 + rbase + r) * HD + col] = (__bf16)(acc3[r] + bc);
    }
    __syncthreads();

    // scores for this chunk (threads 0..63 = waves 0,1)
    float sc = -3e38f;
    if (tid < CHUNK) {
      if (tid < rows) {
        float acc = 0.f;
#pragma unroll 8
        for (int d = 0; d < HD; ++d) acc += qsh[d] * (float)Kl[tid * HD + d];
        sc = acc * 0.125f; // 1/sqrt(64)
      }
      psh[tid] = sc;
    }
    // wave32 shuffle max-reduction over waves 0 and 1
    float mv = sc;
#pragma unroll
    for (int off = 16; off > 0; off >>= 1) mv = fmaxf(mv, __shfl_xor(mv, off, 32));
    if (wave < 2 && lane == 0) red[wave] = mv;
    __syncthreads();
    float m_c = fmaxf(red[0], red[1]);

    float e = 0.f;
    if (tid < CHUNK) {
      e = (tid < rows) ? __expf(psh[tid] - m_c) : 0.f;
      psh[tid] = e;
    }
    float sv = e;
#pragma unroll
    for (int off = 16; off > 0; off >>= 1) sv += __shfl_xor(sv, off, 32);
    if (wave < 2 && lane == 0) red[2 + wave] = sv;
    __syncthreads();

    int idx = (b * HH + h) * NCHUNK + c;
    if (tid == 0) {
      part_m[idx] = m_c;
      part_l[idx] = red[2] + red[3];
    }
    if (tid < HD) {
      float a = 0.f;
#pragma unroll 8
      for (int s = 0; s < CHUNK; ++s) a += psh[s] * (float)Vl[s * HD + tid];
      part_acc[(size_t)idx * HD + tid] = a;
    }
    __syncthreads(); // protect psh/Kl/Vl before next head
  }
}

// ---------------------------------------------------------------------------
// Merge chunk partials (log-sum-exp) -> recalled (B x D)
// ---------------------------------------------------------------------------
__global__ void combine(const float* __restrict__ part_m, const float* __restrict__ part_l,
                        const float* __restrict__ part_acc, float* __restrict__ recalled) {
  int bh = blockIdx.x;
  int d  = threadIdx.x; // 64
  float m = -3e38f;
  for (int cc = 0; cc < NCHUNK; ++cc) m = fmaxf(m, part_m[bh * NCHUNK + cc]);
  float l = 0.f, acc = 0.f;
  for (int cc = 0; cc < NCHUNK; ++cc) {
    float w = __expf(part_m[bh * NCHUNK + cc] - m);
    l   += part_l[bh * NCHUNK + cc] * w;
    acc += part_acc[((size_t)bh * NCHUNK + cc) * HD + d] * w;
  }
  float r = (l > 0.f) ? acc / l : 0.f;
  int b = bh / HH, h = bh % HH;
  recalled[(size_t)b * DD + h * HD + d] = r;
}

// ---------------------------------------------------------------------------
// gate + residual + layernorm
// ---------------------------------------------------------------------------
__global__ void finalize_k(const float* __restrict__ query, const float* __restrict__ gatepre,
                           const float* __restrict__ recalled2, const float* __restrict__ gamma,
                           const float* __restrict__ beta, const int* __restrict__ mcount,
                           float* __restrict__ out) {
  int b = blockIdx.x, tid = threadIdx.x;
  __shared__ float xs[DD];
  __shared__ float rs[256], rs2[256];
  int mc = mcount[0];
  int nvalid = mc < SS ? mc : SS;
  if (nvalid <= 0) {
    for (int i = tid; i < DD; i += 256) out[(size_t)b * DD + i] = query[(size_t)b * DD + i];
    return;
  }
  float s = 0.f, s2 = 0.f;
  for (int i = tid; i < DD; i += 256) {
    float g = 1.f / (1.f + __expf(-gatepre[(size_t)b * DD + i]));
    float x = query[(size_t)b * DD + i] + g * recalled2[(size_t)b * DD + i];
    xs[i] = x; s += x; s2 += x * x;
  }
  rs[tid] = s; rs2[tid] = s2;
  __syncthreads();
  for (int off = 128; off > 0; off >>= 1) {
    if (tid < off) { rs[tid] += rs[tid + off]; rs2[tid] += rs2[tid + off]; }
    __syncthreads();
  }
  float mu  = rs[0] * (1.f / DD);
  float var = rs2[0] * (1.f / DD) - mu * mu;
  float inv = rsqrtf(var + 1e-5f);
  for (int i = tid; i < DD; i += 256)
    out[(size_t)b * DD + i] = (xs[i] - mu) * inv * gamma[i] + beta[i];
}

// ---------------------------------------------------------------------------
extern "C" void kernel_launch(void* const* d_in, const int* in_sizes, int n_in,
                              void* d_out, int out_size, void* d_ws, size_t ws_size,
                              hipStream_t stream) {
  const float* query = (const float*)d_in[0];
  const float* mem   = (const float*)d_in[1];
  const float* Wq    = (const float*)d_in[2];
  const float* bq    = (const float*)d_in[3];
  const float* Wk    = (const float*)d_in[4];
  const float* bk    = (const float*)d_in[5];
  const float* Wv    = (const float*)d_in[6];
  const float* bv    = (const float*)d_in[7];
  const float* Wo    = (const float*)d_in[8];
  const float* bo    = (const float*)d_in[9];
  const float* Wg    = (const float*)d_in[10];
  const float* bg    = (const float*)d_in[11];
  const float* gamma = (const float*)d_in[12];
  const float* beta  = (const float*)d_in[13];
  const int*   mcnt  = (const int*)d_in[14];

  float* ws = (float*)d_ws;
  float* Qws       = ws;                       ws += BB * DD;
  float* recalled  = ws;                       ws += BB * DD;
  float* recalled2 = ws;                       ws += BB * DD;
  float* gatepre   = ws;                       ws += BB * DD;
  float* part_m    = ws;                       ws += BB * HH * NCHUNK;
  float* part_l    = ws;                       ws += BB * HH * NCHUNK;
  float* part_acc  = ws;                       ws += (size_t)BB * HH * NCHUNK * HD;
  __bf16* wkf = (__bf16*)ws;
  __bf16* wvf = wkf + (size_t)DD * DD;

  // 0) Wk/Wv -> fragment-ordered bf16
  prep_wb<<<256, 256, 0, stream>>>(Wk, Wv, wkf, wvf);

  // 1) Q = query @ Wq^T + bq
  gemm_bias<<<dim3(8, 4), 256, 0, stream>>>(query, query, DD, DD, Wq, bq, Qws, DD);

  // 2) fused K/V projection + flash-attention partials
  size_t smem = (size_t)(CHUNK * DD + 2 * CHUNK * HD) * sizeof(__bf16) +
                (size_t)(4 * 64 + 8) * sizeof(float);
  attn_chunk<<<dim3(NCHUNK, BB), 256, smem, stream>>>(mem, Qws, wkf, wvf, bk, bv, mcnt,
                                                      part_m, part_l, part_acc);

  // 3) combine partials -> recalled
  combine<<<BB * HH, HD, 0, stream>>>(part_m, part_l, part_acc, recalled);

  // 4) output projection
  gemm_bias<<<dim3(8, 4), 256, 0, stream>>>(recalled, recalled, DD, DD, Wo, bo, recalled2, DD);

  // 5) gate pre-activation over concat(query, recalled2)
  gemm_bias<<<dim3(8, 4), 256, 0, stream>>>(query, recalled2, DD, 2 * DD, Wg, bg, gatepre, DD);

  // 6) gate + residual + layernorm
  finalize_k<<<BB, 256, 0, stream>>>(query, gatepre, recalled2, gamma, beta, mcnt,
                                     (float*)d_out);
}
